// PhysicsMessagePassingModel_89936615178650
// MI455X (gfx1250) — matrix operand order repaired and checked by reading, not compiled, not run on hardware
//
#include <hip/hip_runtime.h>
#include <hip/hip_bf16.h>

// ---------------- WMMA types & helpers (gfx1250, wave32) ----------------
typedef __bf16 v16bf __attribute__((ext_vector_type(16)));
typedef float  v8f   __attribute__((ext_vector_type(8)));
typedef unsigned int   v4u   __attribute__((ext_vector_type(4)));
typedef unsigned int   v8u   __attribute__((ext_vector_type(8)));
typedef float          v4f   __attribute__((ext_vector_type(4)));
typedef int            v4i   __attribute__((ext_vector_type(4)));

#if defined(__has_builtin)
#if __has_builtin(__builtin_amdgcn_global_load_async_to_lds_b128)
#define HAVE_ASYNC_LDS 1
#endif
#endif

__device__ __forceinline__ unsigned short f2bf(float f) {
    union { float f; unsigned u; } v; v.f = f;
    unsigned u = v.u;
    u += 0x7FFFu + ((u >> 16) & 1u);   // round-to-nearest-even
    return (unsigned short)(u >> 16);
}

__device__ __forceinline__ float leaky(float x) { return x > 0.f ? x : 0.01f * x; }

__device__ __forceinline__ v8f zero8() {
    v8f z = {0.f, 0.f, 0.f, 0.f, 0.f, 0.f, 0.f, 0.f};
    return z;
}

// A-fragment: 16x32 bf16 tile from LDS. ISA 16-bit A layout: lanes 0-15 hold
// M=0..15, K {0..7,16..23}; lanes 16-31 hold M=0..15, K {8..15,24..31}.
__device__ __forceinline__ v16bf load_a_lds(const unsigned short* lds, int rowBase,
                                            int stride, int k0, int lane) {
    int row   = rowBase + (lane & 15);
    int khalf = (lane >> 4) * 8;
    const unsigned short* base = lds + row * stride + k0 + khalf;
    v4u lo = *(const v4u*)base;
    v4u hi = *(const v4u*)(base + 16);
    v8u full = __builtin_shufflevector(lo, hi, 0, 1, 2, 3, 4, 5, 6, 7);
    return __builtin_bit_cast(v16bf, full);
}

// B-fragment from pre-packed bf16 weights: tile = (kb*ntCount + nt).
__device__ __forceinline__ v16bf load_b_packed(const unsigned short* __restrict__ P,
                                               int tile, int lane) {
    const unsigned short* p = P + ((size_t)tile << 9) + (lane << 4);
    v8u full = *(const v8u*)p;
    return __builtin_bit_cast(v16bf, full);
}

__device__ __forceinline__ v8f wmma_bf16(v16bf a, v16bf b, v8f c) {
    return __builtin_amdgcn_wmma_f32_16x16x32_bf16(false, a, false, b, (short)0, c,
                                                   false, false);
}

// 16B global -> LDS copy: async on CDNA5 when available (ASYNCcnt).
__device__ __forceinline__ void copy16_g2lds(const unsigned short* g,
                                             unsigned short* l) {
#if defined(HAVE_ASYNC_LDS)
    __builtin_amdgcn_global_load_async_to_lds_b128(
        (__attribute__((address_space(1))) v4i*)g,
        (__attribute__((address_space(3))) v4i*)l, 0, 0);
#else
    *(v4u*)l = *(const v4u*)g;
#endif
}

__device__ __forceinline__ void wait_async_lds() {
#if defined(HAVE_ASYNC_LDS)
    asm volatile("s_wait_asynccnt 0x0" ::: "memory");
#endif
}

// ---------------- Weight pre-pack: f32 (KxN) -> bf16 B-fragment layout ---
__global__ void pack_weights_kernel(const float* __restrict__ W, int K, int Nout,
                                    int ntCount, int total,
                                    unsigned short* __restrict__ out) {
    for (int o = blockIdx.x * blockDim.x + threadIdx.x; o < total;
         o += gridDim.x * blockDim.x) {
        int tile   = o >> 9;
        int within = o & 511;
        int lane   = within >> 4;
        int i      = within & 15;
        int kb     = tile / ntCount;
        int nt     = tile - kb * ntCount;
        int k      = kb * 32 + ((lane >> 4) * 8) + (i & 7) + ((i >> 3) * 16);
        int col    = nt * 16 + (lane & 15);
        out[o] = (k < K) ? f2bf(W[k * Nout + col]) : (unsigned short)0;
    }
}

// ---------------- Encoder: x (N x 9) -> h (N x 128), leaky on hidden ----
__global__ __launch_bounds__(128) void encoder_kernel(
    const float* __restrict__ x, const float* __restrict__ W1,
    const float* __restrict__ b1, const float* __restrict__ W2,
    const float* __restrict__ b2, float* __restrict__ h,
    unsigned short* __restrict__ hb, int N) {
    __shared__ float sx[9];
    __shared__ float sh[128];
    int n = blockIdx.x;
    if (n >= N) return;
    int j = threadIdx.x;
    if (j < 9) sx[j] = x[n * 9 + j];
    __syncthreads();
    float a = b1[j];
#pragma unroll
    for (int i = 0; i < 9; ++i) a += sx[i] * W1[i * 128 + j];
    sh[j] = leaky(a);
    __syncthreads();
    float o = b2[j];
    for (int k = 0; k < 128; ++k) o += sh[k] * W2[k * 128 + j];
    h[n * 128 + j]  = o;
    hb[n * 128 + j] = f2bf(o);
}

// ---------------- Fused edge message kernel ----------------------------
// 128 edges/block, 128 threads (4 waves x 32-edge M-tiles). mi staged bf16 in
// LDS (async-to-LDS gather); hidden overlaid into the wave's own smi rows.
#define EPB 128
__global__ __launch_bounds__(128) void edge_mp_kernel(
    const unsigned short* __restrict__ hb, const float* __restrict__ edge_attr,
    const int* __restrict__ src, const int* __restrict__ dst,
    const unsigned short* __restrict__ Pp1, const float* __restrict__ bp1,
    const unsigned short* __restrict__ Pp2, const float* __restrict__ bp2,
    const unsigned short* __restrict__ Pv1, const float* __restrict__ bv1,
    const unsigned short* __restrict__ Pv2, const float* __restrict__ bv2,
    float* __restrict__ aggr, int E) {
    __shared__ __align__(16) unsigned short smi[EPB * 288];
    __shared__ int sDst[EPB];
    int ebase = blockIdx.x * EPB;

    {   // destination indices (cached for the atomic scatter)
        int ge = ebase + threadIdx.x;
        int c  = (ge < E) ? ge : (E - 1);
        sDst[threadIdx.x] = dst[c] * 128;
    }
    // Stage h[src]|h[dst]: EPB*32 16B chunks, strided across 128 threads.
    for (int idx = threadIdx.x; idx < EPB * 32; idx += 128) {
        int e = idx >> 5, c = (idx & 31) << 3;  // c = half offset in [0,256)
        int ge = ebase + e;
        int ce = (ge < E) ? ge : (E - 1);
        const unsigned short* sp = (c < 128)
                                       ? (hb + src[ce] * 128 + c)
                                       : (hb + dst[ce] * 128 + (c - 128));
        copy16_g2lds(sp, smi + e * 288 + c);
    }
    // edge_attr (cols 256..257) + zero pad (258..287)
    for (int idx = threadIdx.x; idx < EPB * 32; idx += 128) {
        int e = idx >> 5, f = idx & 31;
        unsigned short v = 0;
        int ge = ebase + e;
        if (f < 2 && ge < E) v = f2bf(edge_attr[ge * 2 + f]);
        smi[e * 288 + 256 + f] = v;
    }
    wait_async_lds();
    __syncthreads();

    int wave = threadIdx.x >> 5, lane = threadIdx.x & 31;
    int wrow = wave * 32;
    int col  = lane & 15;

    // GEMM1: (32 x 288) @ (288 x [64p|64v]) -> 32 x 128 hidden.
    v8f acc[2][8];
#pragma unroll
    for (int m = 0; m < 2; ++m)
#pragma unroll
        for (int t = 0; t < 8; ++t) acc[m][t] = zero8();
    for (int kb = 0; kb < 9; ++kb) {
        v16bf a0 = load_a_lds(smi, wrow, 288, kb * 32, lane);
        v16bf a1 = load_a_lds(smi, wrow + 16, 288, kb * 32, lane);
#pragma unroll
        for (int t = 0; t < 8; ++t) {
            const unsigned short* P = (t < 4) ? Pp1 : Pv1;
            v16bf b = load_b_packed(P, kb * 4 + (t & 3), lane);
            acc[0][t] = wmma_bf16(a0, b, acc[0][t]);
            acc[1][t] = wmma_bf16(a1, b, acc[1][t]);
        }
    }
    // bias + leaky -> hidden overlaid into own smi rows (cols 0..127).
#pragma unroll
    for (int t = 0; t < 8; ++t) {
        const float* bias = (t < 4) ? bp1 : bv1;
        float bv = bias[(t & 3) * 16 + col];
#pragma unroll
        for (int m = 0; m < 2; ++m) {
#pragma unroll
            for (int r = 0; r < 8; ++r) {
                int row = m * 16 + (lane >> 4) * 8 + r;
                smi[(wrow + row) * 288 + t * 16 + col] =
                    f2bf(leaky(acc[m][t][r] + bv));
            }
        }
    }
    asm volatile("s_wait_dscnt 0x0" ::: "memory");

    // GEMM2: p: (32x64)@(64x64), v likewise; no final activation.
    v8f acc2[2][8];
#pragma unroll
    for (int m = 0; m < 2; ++m)
#pragma unroll
        for (int t = 0; t < 8; ++t) acc2[m][t] = zero8();
#pragma unroll
    for (int kb = 0; kb < 2; ++kb) {
        v16bf ap0 = load_a_lds(smi, wrow, 288, kb * 32, lane);
        v16bf ap1 = load_a_lds(smi, wrow + 16, 288, kb * 32, lane);
        v16bf av0 = load_a_lds(smi, wrow, 288, 64 + kb * 32, lane);
        v16bf av1 = load_a_lds(smi, wrow + 16, 288, 64 + kb * 32, lane);
#pragma unroll
        for (int t = 0; t < 8; ++t) {
            const unsigned short* P = (t < 4) ? Pp2 : Pv2;
            v16bf b = load_b_packed(P, kb * 4 + (t & 3), lane);
            acc2[0][t] = wmma_bf16((t < 4) ? ap0 : av0, b, acc2[0][t]);
            acc2[1][t] = wmma_bf16((t < 4) ? ap1 : av1, b, acc2[1][t]);
        }
    }
    float bsum[8];
#pragma unroll
    for (int t = 0; t < 8; ++t) {
        const float* bias = (t < 4) ? bp2 : bv2;
        bsum[t] = bias[(t & 3) * 16 + col];
    }
    // atomic scatter of comb=[p|v] (128 cols) into aggr[dst]
#pragma unroll
    for (int m = 0; m < 2; ++m) {
        int rowbase = wrow + m * 16 + (lane >> 4) * 8;
#pragma unroll
        for (int r = 0; r < 8; ++r) {
            int e  = rowbase + r;
            int ge = ebase + e;
            if (ge < E) {
                int dbase = sDst[e];
#pragma unroll
                for (int t = 0; t < 8; ++t) {
                    int gcol = (t < 4) ? (t * 16 + col) : (64 + (t - 4) * 16 + col);
                    atomicAdd(&aggr[dbase + gcol], acc2[m][t][r] + bsum[t]);
                }
            }
        }
    }
}

// ---------------- Node update kernel (256 -> 128 -> 128, leaky both) ----
#define NPB 128
__global__ __launch_bounds__(128) void node_mp_kernel(
    const unsigned short* __restrict__ hb, const float* __restrict__ aggr,
    const unsigned short* __restrict__ Pn1, const float* __restrict__ bn1,
    const unsigned short* __restrict__ Pn2, const float* __restrict__ bn2,
    float* __restrict__ h_out, unsigned short* __restrict__ hb_out, int N) {
    __shared__ __align__(16) unsigned short sin_[NPB * 256];
    int nbase = blockIdx.x * NPB;

    // Stage concat(h, aggr) as bf16: h rows via async 16B copies; aggr cvt.
    for (int idx = threadIdx.x; idx < NPB * 16; idx += 128) {
        int nl = idx >> 4, c = (idx & 15) << 3;
        int gn = nbase + nl;
        int cn = (gn < N) ? gn : (N - 1);
        copy16_g2lds(hb + cn * 128 + c, sin_ + nl * 256 + c);
    }
    for (int idx = threadIdx.x; idx < NPB * 32; idx += 128) {
        int nl = idx >> 5, c = (idx & 31) << 2;
        int gn = nbase + nl;
        int cn = (gn < N) ? gn : (N - 1);
        v4f a = *(const v4f*)(aggr + cn * 128 + c);
        unsigned short* o = sin_ + nl * 256 + 128 + c;
        o[0] = f2bf(a[0]); o[1] = f2bf(a[1]); o[2] = f2bf(a[2]); o[3] = f2bf(a[3]);
    }
    wait_async_lds();
    __syncthreads();

    int wave = threadIdx.x >> 5, lane = threadIdx.x & 31;
    int wrow = wave * 32;
    int col  = lane & 15;

    v8f acc[2][8];
#pragma unroll
    for (int m = 0; m < 2; ++m)
#pragma unroll
        for (int t = 0; t < 8; ++t) acc[m][t] = zero8();
    for (int kb = 0; kb < 8; ++kb) {
        v16bf a0 = load_a_lds(sin_, wrow, 256, kb * 32, lane);
        v16bf a1 = load_a_lds(sin_, wrow + 16, 256, kb * 32, lane);
#pragma unroll
        for (int t = 0; t < 8; ++t) {
            v16bf b = load_b_packed(Pn1, kb * 8 + t, lane);
            acc[0][t] = wmma_bf16(a0, b, acc[0][t]);
            acc[1][t] = wmma_bf16(a1, b, acc[1][t]);
        }
    }
    // bias + leaky -> hidden overlaid into own sin_ rows (cols 0..127)
#pragma unroll
    for (int t = 0; t < 8; ++t) {
        float bv = bn1[t * 16 + col];
#pragma unroll
        for (int m = 0; m < 2; ++m) {
#pragma unroll
            for (int r = 0; r < 8; ++r) {
                int row = m * 16 + (lane >> 4) * 8 + r;
                sin_[(wrow + row) * 256 + t * 16 + col] =
                    f2bf(leaky(acc[m][t][r] + bv));
            }
        }
    }
    asm volatile("s_wait_dscnt 0x0" ::: "memory");

    v8f acc2[2][8];
#pragma unroll
    for (int m = 0; m < 2; ++m)
#pragma unroll
        for (int t = 0; t < 8; ++t) acc2[m][t] = zero8();
#pragma unroll
    for (int kb = 0; kb < 4; ++kb) {
        v16bf a0 = load_a_lds(sin_, wrow, 256, kb * 32, lane);
        v16bf a1 = load_a_lds(sin_, wrow + 16, 256, kb * 32, lane);
#pragma unroll
        for (int t = 0; t < 8; ++t) {
            v16bf b = load_b_packed(Pn2, kb * 8 + t, lane);
            acc2[0][t] = wmma_bf16(a0, b, acc2[0][t]);
            acc2[1][t] = wmma_bf16(a1, b, acc2[1][t]);
        }
    }
#pragma unroll
    for (int t = 0; t < 8; ++t) {
        float bv = bn2[t * 16 + col];
#pragma unroll
        for (int m = 0; m < 2; ++m) {
#pragma unroll
            for (int r = 0; r < 8; ++r) {
                int row = m * 16 + (lane >> 4) * 8 + r;
                int gn  = nbase + wrow + row;
                if (gn < N) {
                    float v = leaky(acc2[m][t][r] + bv);
                    h_out[gn * 128 + t * 16 + col]  = v;
                    hb_out[gn * 128 + t * 16 + col] = f2bf(v);
                }
            }
        }
    }
}

// ---------------- Decoder + equality correction (scalar tail) -----------
__global__ __launch_bounds__(128) void decode_kernel(
    const float* __restrict__ h, const float* __restrict__ x,
    const float* __restrict__ Wd1, const float* __restrict__ bd1,
    const float* __restrict__ Wd2, const float* __restrict__ bd2,
    const float* __restrict__ Wq1, const float* __restrict__ bq1,
    const float* __restrict__ Wq2, const float* __restrict__ bq2,
    float* __restrict__ pred, int N) {
    __shared__ float sh[128], shd[128], she[64];
    int n = blockIdx.x;
    if (n >= N) return;
    int j = threadIdx.x;
    sh[j] = h[n * 128 + j];
    __syncthreads();
    float a = bd1[j];
    for (int k = 0; k < 128; ++k) a += sh[k] * Wd1[k * 128 + j];
    shd[j] = leaky(a);
    if (j < 64) {
        float e = bq1[j];
        for (int k = 0; k < 128; ++k) e += sh[k] * Wq1[k * 64 + j];
        she[j] = leaky(e);
    }
    __syncthreads();
    if (j < 6) {
        float p = bd2[j];
        for (int k = 0; k < 128; ++k) p += shd[k] * Wd2[k * 6 + j];
        float c = bq2[j];
        for (int k = 0; k < 64; ++k) c += she[k] * Wq2[k * 6 + j];
        float pv = x[n * 9 + 7], ref = x[n * 9 + 8];
        if (j == 0 && (pv == 1.f || ref == 1.f)) c = 0.f;  // VM
        if (j == 1 && ref == 1.f) c = 0.f;                 // VA
        pred[n * 6 + j] = p - 3.f * c;                     // three corrections
    }
}

// ---------------- Power-flow edge kernel --------------------------------
__global__ void pf_edge_kernel(const float* __restrict__ pred,
                               const float* __restrict__ edge_attr,
                               const int* __restrict__ src,
                               const int* __restrict__ dst,
                               float* __restrict__ pcalc,
                               float* __restrict__ qcalc, int E) {
    int e = blockIdx.x * blockDim.x + threadIdx.x;
    if (e >= E) return;
    int s = src[e], d = dst[e];
    float g = edge_attr[e * 2 + 0], b = edge_attr[e * 2 + 1];
    float vf = pred[s * 6 + 0], vt = pred[d * 6 + 0];
    float ad = pred[s * 6 + 1] - pred[d * 6 + 1];
    float c = cosf(ad), sn = sinf(ad);
    float vv = vf * vt;
    atomicAdd(&pcalc[s], vv * (g * c + b * sn));
    atomicAdd(&qcalc[s], vv * (g * sn - b * c));
}

// ---------------- Final correction --------------------------------------
__global__ void final_kernel(const float* __restrict__ pred,
                             const float* __restrict__ x,
                             const float* __restrict__ pcalc,
                             const float* __restrict__ qcalc,
                             float* __restrict__ out, int N) {
    int n = blockIdx.x * blockDim.x + threadIdx.x;
    if (n >= N) return;
    float p[6];
#pragma unroll
    for (int c = 0; c < 6; ++c) p[c] = pred[n * 6 + c];
    float imbp = p[2] - p[4] - pcalc[n];
    float imbq = p[3] - p[5] - qcalc[n];
    float pq = x[n * 9 + 6], pv = x[n * 9 + 7];
    if (pq == 1.f) p[0] += imbq * 0.05f;
    if (pv == 1.f) p[3] += imbq * 0.05f;
    p[2] += imbp * 0.05f;
#pragma unroll
    for (int c = 0; c < 6; ++c) out[n * 6 + c] = p[c];
}

// ---------------- Host launch -------------------------------------------
extern "C" void kernel_launch(void* const* d_in, const int* in_sizes, int n_in,
                              void* d_out, int out_size, void* d_ws, size_t ws_size,
                              hipStream_t stream) {
    const int N = in_sizes[0] / 9;
    const int E = in_sizes[2] / 2;

    const float* x         = (const float*)d_in[0];
    const float* edge_attr = (const float*)d_in[2];
    auto f = [&](int i) { return (const float*)d_in[i]; };
    const int P = 3;  // params leaves: dec(4), enc(4), eq(4), 5x[node,p,v](12)
    const float *Wd1 = f(P + 0), *bd1 = f(P + 1), *Wd2 = f(P + 2), *bd2 = f(P + 3);
    const float *We1 = f(P + 4), *be1 = f(P + 5), *We2 = f(P + 6), *be2 = f(P + 7);
    const float *Wq1 = f(P + 8), *bq1 = f(P + 9), *Wq2 = f(P + 10), *bq2 = f(P + 11);
    struct LP {
        const float *Wn1, *bn1, *Wn2, *bn2, *Wp1, *bp1, *Wp2, *bp2, *Wv1, *bv1, *Wv2, *bv2;
    } L[5];
    for (int l = 0; l < 5; ++l) {
        int b = P + 12 + l * 12;
        L[l] = {f(b + 0), f(b + 1), f(b + 2), f(b + 3), f(b + 4),  f(b + 5),
                f(b + 6), f(b + 7), f(b + 8), f(b + 9), f(b + 10), f(b + 11)};
    }
    const int* edge_index = (const int*)d_in[P + 12 + 60];
    const int* src = edge_index;
    const int* dst = edge_index + E;

    // Workspace carve-up
    char* w = (char*)d_ws;
    auto carve = [&](size_t bytes) {
        char* p = w;
        w += (bytes + 255) & ~(size_t)255;
        return p;
    };
    float* h0           = (float*)carve((size_t)N * 128 * 4);
    float* h1           = (float*)carve((size_t)N * 128 * 4);
    unsigned short* hb0 = (unsigned short*)carve((size_t)N * 128 * 2);
    unsigned short* hb1 = (unsigned short*)carve((size_t)N * 128 * 2);
    float* aggr         = (float*)carve((size_t)N * 128 * 4);
    float* pred         = (float*)carve((size_t)N * 6 * 4);
    float* pcalc        = (float*)carve((size_t)N * 4);
    float* qcalc        = (float*)carve((size_t)N * 4);

    const int SZ_P1 = 9 * 4 * 512, SZ_P2 = 2 * 4 * 512;
    const int SZ_N1 = 8 * 8 * 512, SZ_N2 = 4 * 8 * 512;
    unsigned short *Pp1[5], *Pv1[5], *Pp2[5], *Pv2[5], *Pn1[5], *Pn2[5];
    for (int l = 0; l < 5; ++l) {
        Pp1[l] = (unsigned short*)carve((size_t)SZ_P1 * 2);
        Pv1[l] = (unsigned short*)carve((size_t)SZ_P1 * 2);
        Pp2[l] = (unsigned short*)carve((size_t)SZ_P2 * 2);
        Pv2[l] = (unsigned short*)carve((size_t)SZ_P2 * 2);
        Pn1[l] = (unsigned short*)carve((size_t)SZ_N1 * 2);
        Pn2[l] = (unsigned short*)carve((size_t)SZ_N2 * 2);
    }
    for (int l = 0; l < 5; ++l) {
        pack_weights_kernel<<<(SZ_P1 + 255) / 256, 256, 0, stream>>>(
            L[l].Wp1, 258, 64, 4, SZ_P1, Pp1[l]);
        pack_weights_kernel<<<(SZ_P1 + 255) / 256, 256, 0, stream>>>(
            L[l].Wv1, 258, 64, 4, SZ_P1, Pv1[l]);
        pack_weights_kernel<<<(SZ_P2 + 255) / 256, 256, 0, stream>>>(
            L[l].Wp2, 64, 64, 4, SZ_P2, Pp2[l]);
        pack_weights_kernel<<<(SZ_P2 + 255) / 256, 256, 0, stream>>>(
            L[l].Wv2, 64, 64, 4, SZ_P2, Pv2[l]);
        pack_weights_kernel<<<(SZ_N1 + 255) / 256, 256, 0, stream>>>(
            L[l].Wn1, 256, 128, 8, SZ_N1, Pn1[l]);
        pack_weights_kernel<<<(SZ_N2 + 255) / 256, 256, 0, stream>>>(
            L[l].Wn2, 128, 128, 8, SZ_N2, Pn2[l]);
    }

    encoder_kernel<<<N, 128, 0, stream>>>(x, We1, be1, We2, be2, h0, hb0, N);

    float* hc = h0;
    unsigned short* hbc = hb0;
    float* hn = h1;
    unsigned short* hbn = hb1;
    for (int l = 0; l < 5; ++l) {
        (void)hipMemsetAsync(aggr, 0, (size_t)N * 128 * 4, stream);
        edge_mp_kernel<<<(E + EPB - 1) / EPB, 128, 0, stream>>>(
            hbc, edge_attr, src, dst, Pp1[l], L[l].bp1, Pp2[l], L[l].bp2, Pv1[l],
            L[l].bv1, Pv2[l], L[l].bv2, aggr, E);
        node_mp_kernel<<<(N + NPB - 1) / NPB, 128, 0, stream>>>(
            hbc, aggr, Pn1[l], L[l].bn1, Pn2[l], L[l].bn2, hn, hbn, N);
        float* tf = hc; hc = hn; hn = tf;
        unsigned short* tb = hbc; hbc = hbn; hbn = tb;
    }

    decode_kernel<<<N, 128, 0, stream>>>(hc, x, Wd1, bd1, Wd2, bd2, Wq1, bq1, Wq2,
                                         bq2, pred, N);
    (void)hipMemsetAsync(pcalc, 0, (size_t)N * 4, stream);
    (void)hipMemsetAsync(qcalc, 0, (size_t)N * 4, stream);
    pf_edge_kernel<<<(E + 255) / 256, 256, 0, stream>>>(pred, edge_attr, src, dst,
                                                        pcalc, qcalc, E);
    final_kernel<<<(N + 255) / 256, 256, 0, stream>>>(pred, x, pcalc, qcalc,
                                                      (float*)d_out, N);
}
